// SeqLinear_7275674599456
// MI455X (gfx1250) — matrix operand order repaired and checked by reading, not compiled, not run on hardware
//
#include <hip/hip_runtime.h>
#include <hip/hip_bf16.h>

// ---------------------------------------------------------------------------
// Mamba-2 SSD forward for MI455X (gfx1250), wave32 + WMMA bf16.
// ---------------------------------------------------------------------------

typedef __bf16 bf16_t;
typedef __attribute__((ext_vector_type(16))) __bf16 v16bf;
typedef __attribute__((ext_vector_type(8)))  float  v8f;

#define SEQ     4096
#define NB      4
#define NHEAD   16
#define DCONVC  3072
#define DPROJ   3088
#define LDP     3136   // padded proj leading dim (multiple of 64)
#define NCHUNK  64
#define TSTR    80     // LDS tile stride (64 + 16 pad), 160B rows (16B aligned)

// ---- WMMA helpers ---------------------------------------------------------

__device__ __forceinline__ v8f wmma_bf16(v16bf a, v16bf b, v8f c) {
  return __builtin_amdgcn_wmma_f32_16x16x32_bf16(false, a, false, b,
                                                 (short)0, c, false, false);
}

// A operand (16x32 bf16): lane holds row M = rowBase + (lane&15);
// lanes 16-31 shift K base by 8. Elements 0..7 = K kh..kh+7, 8..15 = kh+16..kh+23.
__device__ __forceinline__ v16bf load_frag_a(const bf16_t* lds, int rowBase,
                                             int stride, int k0, int lane) {
  int row = rowBase + (lane & 15);
  int kh  = (lane & 16) ? 8 : 0;
  const bf16_t* p = lds + row * stride + k0 + kh;
  v16bf f;
  reinterpret_cast<uint4*>(&f)[0] = *reinterpret_cast<const uint4*>(p);
  reinterpret_cast<uint4*>(&f)[1] = *reinterpret_cast<const uint4*>(p + 16);
  return f;
}

// B operand (32x16 bf16, stored transposed as N x K row-major):
// lane holds row N = rowBase + (lane&15); lanes 16-31 shift K base by 8.
// Elements 0..7 = K kh..kh+7, 8..15 = kh+16..kh+23.
__device__ __forceinline__ v16bf load_frag_b(const bf16_t* lds, int rowBase,
                                             int stride, int k0, int lane) {
  int row = rowBase + (lane & 15);
  int kh  = (lane & 16) ? 8 : 0;
  const bf16_t* p = lds + row * stride + k0 + kh;
  v16bf f;
  reinterpret_cast<uint4*>(&f)[0] = *reinterpret_cast<const uint4*>(p);
  reinterpret_cast<uint4*>(&f)[1] = *reinterpret_cast<const uint4*>(p + 16);
  return f;
}

// ---- elementwise fp32 -> bf16 ---------------------------------------------

__global__ void cvt_kernel(const float* __restrict__ src,
                           bf16_t* __restrict__ dst, size_t n) {
  size_t i = (size_t)blockIdx.x * 256 + threadIdx.x;
  if (i < n) dst[i] = (bf16_t)src[i];
}

// ---- WMMA GEMM: C(MxN,f32) = A(MxK,bf16 rowmajor) * B(NxK,bf16 rowmajor)^T
// Block tile 128x128, BK=64, 256 threads = 8 waves (2x4 wave grid),
// wave tile 64x32 = 4x2 wmma accumulators, 16 WMMA per wave per LDS refresh.
// Requires M % 128 == 0, K % 64 == 0. N guarded.

__global__ __launch_bounds__(256) void gemm_bf16_wmma(
    const bf16_t* __restrict__ A, const bf16_t* __restrict__ B,
    float* __restrict__ C, int M, int N, int K, int ldc) {
  __shared__ __attribute__((aligned(16))) bf16_t As[128 * TSTR];
  __shared__ __attribute__((aligned(16))) bf16_t Bs[128 * TSTR];

  const int tid  = threadIdx.x;
  const int lane = tid & 31;
  const int w    = tid >> 5;
  const int wr   = w & 1;      // 2 row groups of 64
  const int wc   = w >> 1;     // 4 col groups of 32
  const int mBase = blockIdx.x * 128;
  const int nBase = blockIdx.y * 128;

  const v8f vzero = {0.f,0.f,0.f,0.f,0.f,0.f,0.f,0.f};
  v8f acc[4][2];
#pragma unroll
  for (int i = 0; i < 4; ++i)
#pragma unroll
    for (int j = 0; j < 2; ++j) acc[i][j] = vzero;

  const int nk = K >> 6;
  for (int kb = 0; kb < nk; ++kb) {
    const int k0 = kb << 6;
    // A tile: 128 rows x 64 cols = 1024 x 16B chunks, 4 per thread
#pragma unroll
    for (int q = 0; q < 4; ++q) {
      int cidx = tid + q * 256;
      int row = cidx >> 3, seg = cidx & 7;
      uint4 v = *reinterpret_cast<const uint4*>(
          A + (size_t)(mBase + row) * K + k0 + seg * 8);
      *reinterpret_cast<uint4*>(&As[row * TSTR + seg * 8]) = v;
    }
    // B tile: 128 rows x 64 cols, N-guarded
#pragma unroll
    for (int q = 0; q < 4; ++q) {
      int cidx = tid + q * 256;
      int row = cidx >> 3, seg = cidx & 7;
      uint4 v{};
      int n = nBase + row;
      if (n < N)
        v = *reinterpret_cast<const uint4*>(B + (size_t)n * K + k0 + seg * 8);
      *reinterpret_cast<uint4*>(&Bs[row * TSTR + seg * 8]) = v;
    }
    if (kb + 1 < nk) {  // near-cache prefetch of next A tile chunk
      int row = tid >> 3, seg = tid & 7;
      __builtin_prefetch(A + (size_t)(mBase + row) * K + k0 + 64 + seg * 8, 0, 3);
    }
    __syncthreads();

#pragma unroll
    for (int ks = 0; ks < 2; ++ks) {
      const int kk = ks * 32;
      v16bf bfr[2], afr[4];
#pragma unroll
      for (int j = 0; j < 2; ++j)
        bfr[j] = load_frag_b(Bs, wc * 32 + j * 16, TSTR, kk, lane);
#pragma unroll
      for (int i = 0; i < 4; ++i)
        afr[i] = load_frag_a(As, wr * 64 + i * 16, TSTR, kk, lane);
#pragma unroll
      for (int i = 0; i < 4; ++i)
#pragma unroll
        for (int j = 0; j < 2; ++j)
          acc[i][j] = wmma_bf16(afr[i], bfr[j], acc[i][j]);
    }
    __syncthreads();
  }

  const int colN   = lane & 15;
  const int rowOff = (lane & 16) ? 8 : 0;
#pragma unroll
  for (int i = 0; i < 4; ++i)
#pragma unroll
    for (int j = 0; j < 2; ++j) {
      int col = nBase + wc * 32 + j * 16 + colN;
      if (col < N) {
#pragma unroll
        for (int r = 0; r < 8; ++r) {
          int rowg = mBase + wr * 64 + i * 16 + rowOff + r;
          C[(size_t)rowg * ldc + col] = acc[i][j][r];
        }
      }
    }
}

// ---- depthwise causal conv (b=0 only) -> head-major bf16 C/B/x ------------

__global__ void conv_kernel(const float* __restrict__ proj,
                            const float* __restrict__ conv_w,
                            const float* __restrict__ conv_b,
                            bf16_t* __restrict__ Cbf, bf16_t* __restrict__ Bbf,
                            bf16_t* __restrict__ Xbf) {
  int tid = blockIdx.x * 256 + threadIdx.x;
  if (tid >= SEQ * DCONVC) return;
  int t = tid / DCONVC, ch = tid % DCONVC;
  float acc = conv_b[ch];
#pragma unroll
  for (int k = 0; k < 4; ++k) {
    int tt = t - 3 + k;
    if (tt >= 0) acc += proj[(size_t)tt * LDP + ch] * conv_w[ch * 4 + k];
  }
  bf16_t v = (bf16_t)acc;
  if (ch < 1024) {
    int h = ch >> 6;
    Cbf[((size_t)h * SEQ + t) * 64 + (ch & 63)] = v;
  } else if (ch < 2048) {
    int c2 = ch - 1024, h = c2 >> 6;
    Bbf[((size_t)h * SEQ + t) * 64 + (c2 & 63)] = v;
  } else {
    int c2 = ch - 2048, h = c2 >> 6;
    Xbf[((size_t)h * SEQ + t) * 64 + (c2 & 63)] = v;
  }
}

// ---- dt -> softplus -> A = A_param * dt, layout (b,h,s) -------------------

__global__ void dtA_kernel(const float* __restrict__ proj,
                           const float* __restrict__ A_param,
                           const float* __restrict__ dt_bias,
                           float* __restrict__ Abuf) {
  int tid = blockIdx.x * 256 + threadIdx.x;
  if (tid >= NB * SEQ * NHEAD) return;
  int b = tid / (SEQ * NHEAD);
  int r = tid % (SEQ * NHEAD);
  int s = r / NHEAD, h = r % NHEAD;
  float v = proj[((size_t)b * SEQ + s) * LDP + DCONVC + h] + dt_bias[h];
  float sp = (v > 20.f) ? v : log1pf(__expf(v));
  Abuf[((size_t)b * NHEAD + h) * SEQ + s] = A_param[h] * sp;
}

// ---- serial cumsum + norm per (b,h) ---------------------------------------

__global__ void cum_kernel(const float* __restrict__ Abuf,
                           float* __restrict__ cumA,
                           float* __restrict__ normB) {
  int g = blockIdx.x * 32 + threadIdx.x;
  if (g >= NB * NHEAD) return;
  size_t base = (size_t)g * SEQ;
  float ca = 0.f, acc = 0.f;
  for (int t = 0; t < SEQ; ++t) {
    ca += Abuf[base + t];
    cumA[base + t] = ca;
    acc += __expf(-ca);
    normB[base + t] = __expf(ca) * acc;
  }
}

// ---- SSD per-(chunk,head) kernel: Y_diag + per-chunk states (b=0) ---------
// G = C*B^T ; G' = tril(G) * exp(cumA[l]-cumA[s]) ; Y_diag = G' * x
// states[p,n] = sum_s x[s,p]*exp(caEnd-cumA[s])*B[s,n]

__global__ __launch_bounds__(256) void ssd_chunk_kernel(
    const bf16_t* __restrict__ Cbf, const bf16_t* __restrict__ Bbf,
    const bf16_t* __restrict__ Xbf, const float* __restrict__ cumA,
    float* __restrict__ Ydiag, float* __restrict__ Sraw) {
  __shared__ __attribute__((aligned(16))) bf16_t Cs[64 * TSTR];
  __shared__ __attribute__((aligned(16))) bf16_t Bsm[64 * TSTR];
  __shared__ __attribute__((aligned(16))) bf16_t Xt[64 * TSTR];   // x^T (p,s)
  __shared__ __attribute__((aligned(16))) bf16_t Xtd[64 * TSTR];  // decayed x^T
  __shared__ __attribute__((aligned(16))) bf16_t Bt[64 * TSTR];   // B^T (n,s)
  __shared__ __attribute__((aligned(16))) bf16_t Gs[64 * TSTR];   // masked G (l,s)
  __shared__ float caL[64];

  const int c = blockIdx.x, h = blockIdx.y;
  const int tid = threadIdx.x, lane = tid & 31, w = tid >> 5;
  const size_t hbase = (size_t)h * SEQ;
  const int s0 = c * 64;
  const float* cumAh = cumA + (size_t)h * SEQ;  // b = 0 slice
  const float caEnd = cumAh[s0 + 63];

  // C,B row-major tiles: 64 rows x 128B = 512 x 16B chunks, 2 per thread
#pragma unroll
  for (int q = 0; q < 2; ++q) {
    int cidx = tid + q * 256;
    int row = cidx >> 3, seg = cidx & 7;
    *reinterpret_cast<uint4*>(&Cs[row * TSTR + seg * 8]) =
        *reinterpret_cast<const uint4*>(Cbf + (hbase + s0 + row) * 64 + seg * 8);
    *reinterpret_cast<uint4*>(&Bsm[row * TSTR + seg * 8]) =
        *reinterpret_cast<const uint4*>(Bbf + (hbase + s0 + row) * 64 + seg * 8);
  }
  // transposed tiles (coalesced reads, scattered b16 LDS writes)
  {
    int s = tid >> 2;
    int p0 = (tid & 3) * 16;
    float d = __expf(caEnd - cumAh[s0 + s]);
    const bf16_t* xr = Xbf + (hbase + s0 + s) * 64 + p0;
    const bf16_t* br = Bbf + (hbase + s0 + s) * 64 + p0;
#pragma unroll
    for (int i = 0; i < 16; ++i) {
      bf16_t xv = xr[i];
      Xt[(p0 + i) * TSTR + s]  = xv;
      Xtd[(p0 + i) * TSTR + s] = (bf16_t)((float)xv * d);
      Bt[(p0 + i) * TSTR + s]  = br[i];
    }
  }
  if (tid < 64) caL[tid] = cumAh[s0 + tid];
  __syncthreads();

  const int colN   = lane & 15;
  const int rowOff = (lane & 16) ? 8 : 0;
  const v8f vzero = {0.f,0.f,0.f,0.f,0.f,0.f,0.f,0.f};

  // Phase 1: G = C @ B^T, mask+scale, store bf16 into Gs
#pragma unroll
  for (int i = 0; i < 2; ++i) {
    int T = w * 2 + i, tm = T >> 2, tn = T & 3;
    v8f g = vzero;
    g = wmma_bf16(load_frag_a(Cs, tm * 16, TSTR, 0, lane),
                  load_frag_b(Bsm, tn * 16, TSTR, 0, lane), g);
    g = wmma_bf16(load_frag_a(Cs, tm * 16, TSTR, 32, lane),
                  load_frag_b(Bsm, tn * 16, TSTR, 32, lane), g);
#pragma unroll
    for (int r = 0; r < 8; ++r) {
      int l = tm * 16 + rowOff + r, s = tn * 16 + colN;
      float v = (s <= l) ? g[r] * __expf(caL[l] - caL[s]) : 0.f;
      Gs[l * TSTR + s] = (bf16_t)v;
    }
  }
  __syncthreads();

  const size_t obase = ((size_t)c * NHEAD + h) * 4096;
#pragma unroll
  for (int i = 0; i < 2; ++i) {
    int T = w * 2 + i, tm = T >> 2, tn = T & 3;
    // Phase 2: Y_diag(l,p) = Gs(l,s) @ Xt(p,s)^T
    v8f yd = vzero;
    yd = wmma_bf16(load_frag_a(Gs, tm * 16, TSTR, 0, lane),
                   load_frag_b(Xt, tn * 16, TSTR, 0, lane), yd);
    yd = wmma_bf16(load_frag_a(Gs, tm * 16, TSTR, 32, lane),
                   load_frag_b(Xt, tn * 16, TSTR, 32, lane), yd);
#pragma unroll
    for (int r = 0; r < 8; ++r) {
      int l = tm * 16 + rowOff + r, p = tn * 16 + colN;
      Ydiag[obase + l * 64 + p] = yd[r];
    }
    // Phase 3: states(p,n) = Xtd(p,s) @ Bt(n,s)^T
    v8f st = vzero;
    st = wmma_bf16(load_frag_a(Xtd, tm * 16, TSTR, 0, lane),
                   load_frag_b(Bt, tn * 16, TSTR, 0, lane), st);
    st = wmma_bf16(load_frag_a(Xtd, tm * 16, TSTR, 32, lane),
                   load_frag_b(Bt, tn * 16, TSTR, 32, lane), st);
#pragma unroll
    for (int r = 0; r < 8; ++r) {
      int p = tm * 16 + rowOff + r, n = tn * 16 + colN;
      Sraw[obase + p * 64 + n] = st[r];
    }
  }
}

// ---- inter-chunk decay scan over chunks (b=0, per head) -------------------
// used[c] = S ; S = exp(chunkSum_c)*S + states_c

__global__ __launch_bounds__(256) void scan_kernel(
    const float* __restrict__ Sraw, bf16_t* __restrict__ Sused,
    const float* __restrict__ cumA) {
  int h = blockIdx.x, tid = threadIdx.x;
  const float* cumAh = cumA + (size_t)h * SEQ;
  float S[16];
#pragma unroll
  for (int i = 0; i < 16; ++i) S[i] = 0.f;
  int e0 = tid * 16;
  for (int c = 0; c < NCHUNK; ++c) {
    size_t base = ((size_t)c * NHEAD + h) * 4096 + e0;
    float dA = cumAh[c * 64 + 63] - (c ? cumAh[c * 64 - 1] : 0.f);
    float wd = __expf(dA);
#pragma unroll
    for (int i = 0; i < 16; ++i) {
      float raw = Sraw[base + i];
      Sused[base + i] = (bf16_t)S[i];
      S[i] = wd * S[i] + raw;
    }
  }
}

// ---- Y_off = (C @ states_used^T) * exp(in-chunk cumA), added into Y -------

__global__ __launch_bounds__(256) void yoff_kernel(
    const bf16_t* __restrict__ Cbf, const bf16_t* __restrict__ Sused,
    const float* __restrict__ cumA, float* __restrict__ Y) {
  __shared__ __attribute__((aligned(16))) bf16_t Cs[64 * TSTR];
  __shared__ __attribute__((aligned(16))) bf16_t Ss[64 * TSTR];  // (p,n)
  __shared__ float caL[64];

  const int c = blockIdx.x, h = blockIdx.y;
  const int tid = threadIdx.x, lane = tid & 31, w = tid >> 5;
  const size_t hbase = (size_t)h * SEQ;
  const int s0 = c * 64;
  const float* cumAh = cumA + (size_t)h * SEQ;
  const float caPrev = (c == 0) ? 0.f : cumAh[s0 - 1];
  const size_t obase = ((size_t)c * NHEAD + h) * 4096;

#pragma unroll
  for (int q = 0; q < 2; ++q) {
    int cidx = tid + q * 256;
    int row = cidx >> 3, seg = cidx & 7;
    *reinterpret_cast<uint4*>(&Cs[row * TSTR + seg * 8]) =
        *reinterpret_cast<const uint4*>(Cbf + (hbase + s0 + row) * 64 + seg * 8);
    *reinterpret_cast<uint4*>(&Ss[row * TSTR + seg * 8]) =
        *reinterpret_cast<const uint4*>(Sused + obase + row * 64 + seg * 8);
  }
  if (tid < 64) caL[tid] = cumAh[s0 + tid];
  __syncthreads();

  const int colN   = lane & 15;
  const int rowOff = (lane & 16) ? 8 : 0;
  const v8f vzero = {0.f,0.f,0.f,0.f,0.f,0.f,0.f,0.f};
#pragma unroll
  for (int i = 0; i < 2; ++i) {
    int T = w * 2 + i, tm = T >> 2, tn = T & 3;
    v8f yo = vzero;
    yo = wmma_bf16(load_frag_a(Cs, tm * 16, TSTR, 0, lane),
                   load_frag_b(Ss, tn * 16, TSTR, 0, lane), yo);
    yo = wmma_bf16(load_frag_a(Cs, tm * 16, TSTR, 32, lane),
                   load_frag_b(Ss, tn * 16, TSTR, 32, lane), yo);
#pragma unroll
    for (int r = 0; r < 8; ++r) {
      int l = tm * 16 + rowOff + r, p = tn * 16 + colN;
      size_t idx = obase + l * 64 + p;
      Y[idx] = Y[idx] + yo[r] * __expf(caL[l] - caPrev);
    }
  }
}

// ---- Z[b,s,hp] = bf16( Y0[s,h,p] / norm[b,s,h] ) --------------------------

__global__ void z_kernel(const float* __restrict__ Y,
                         const float* __restrict__ normB,
                         bf16_t* __restrict__ Z) {
  size_t tid = (size_t)blockIdx.x * 256 + threadIdx.x;
  if (tid >= (size_t)NB * SEQ * 1024) return;
  int b = (int)(tid / ((size_t)SEQ * 1024));
  int r = (int)(tid % ((size_t)SEQ * 1024));
  int s = r / 1024, ch = r % 1024;
  int h = ch >> 6, p = ch & 63;
  int c = s >> 6, l = s & 63;
  float y = Y[(((size_t)c * NHEAD + h) * 4096) + l * 64 + p];
  float nm = normB[((size_t)b * NHEAD + h) * SEQ + s];
  Z[tid] = (bf16_t)(y / nm);
}

// ---------------------------------------------------------------------------

extern "C" void kernel_launch(void* const* d_in, const int* in_sizes, int n_in,
                              void* d_out, int out_size, void* d_ws,
                              size_t ws_size, hipStream_t stream) {
  const float* x       = (const float*)d_in[0];  // (4,4096,1024)
  const float* W_in    = (const float*)d_in[1];  // (3088,1024)
  const float* conv_w  = (const float*)d_in[2];  // (3072,4)
  const float* conv_b  = (const float*)d_in[3];  // (3072,)
  const float* A_param = (const float*)d_in[4];  // (16,)
  const float* dt_bias = (const float*)d_in[5];  // (16,)
  const float* W_out   = (const float*)d_in[6];  // (1024,1024)
  float* out = (float*)d_out;                    // (4,4096,1024) fp32
  (void)in_sizes; (void)n_in; (void)out_size; (void)ws_size;

  char* p = (char*)d_ws;
  auto alloc = [&](size_t bytes) -> char* {
    char* r = p;
    p += (bytes + 255) & ~(size_t)255;
    return r;
  };
  bf16_t* Xbf16  = (bf16_t*)alloc((size_t)NB * SEQ * 1024 * 2);
  bf16_t* WinBf  = (bf16_t*)alloc((size_t)DPROJ * 1024 * 2);
  bf16_t* WoutBf = (bf16_t*)alloc((size_t)1024 * 1024 * 2);
  float*  Proj   = (float*)alloc((size_t)NB * SEQ * LDP * 4);
  bf16_t* Ch_bf  = (bf16_t*)alloc((size_t)NHEAD * SEQ * 64 * 2);
  bf16_t* Bh_bf  = (bf16_t*)alloc((size_t)NHEAD * SEQ * 64 * 2);
  bf16_t* Xh_bf  = (bf16_t*)alloc((size_t)NHEAD * SEQ * 64 * 2);
  float*  Abuf   = (float*)alloc((size_t)NB * NHEAD * SEQ * 4);
  float*  CumA   = (float*)alloc((size_t)NB * NHEAD * SEQ * 4);
  float*  NormB  = (float*)alloc((size_t)NB * NHEAD * SEQ * 4);
  float*  Sraw   = (float*)alloc((size_t)NCHUNK * NHEAD * 4096 * 4);
  bf16_t* Sused  = (bf16_t*)alloc((size_t)NCHUNK * NHEAD * 4096 * 2);
  float*  Ybuf   = (float*)alloc((size_t)NCHUNK * NHEAD * 4096 * 4);
  bf16_t* Zbf    = (bf16_t*)alloc((size_t)NB * SEQ * 1024 * 2);

  const size_t nX = (size_t)NB * SEQ * 1024;
  cvt_kernel<<<(unsigned)((nX + 255) / 256), 256, 0, stream>>>(x, Xbf16, nX);
  const size_t nWi = (size_t)DPROJ * 1024;
  cvt_kernel<<<(unsigned)((nWi + 255) / 256), 256, 0, stream>>>(W_in, WinBf, nWi);
  const size_t nWo = (size_t)1024 * 1024;
  cvt_kernel<<<(unsigned)((nWo + 255) / 256), 256, 0, stream>>>(W_out, WoutBf, nWo);

  // proj = X @ W_in^T  (M=16384, N=3088, K=1024)
  gemm_bf16_wmma<<<dim3(NB * SEQ / 128, (DPROJ + 127) / 128), 256, 0, stream>>>(
      Xbf16, WinBf, Proj, NB * SEQ, DPROJ, 1024, LDP);

  conv_kernel<<<(SEQ * DCONVC + 255) / 256, 256, 0, stream>>>(
      Proj, conv_w, conv_b, Ch_bf, Bh_bf, Xh_bf);
  dtA_kernel<<<(NB * SEQ * NHEAD + 255) / 256, 256, 0, stream>>>(
      Proj, A_param, dt_bias, Abuf);
  cum_kernel<<<2, 32, 0, stream>>>(Abuf, CumA, NormB);

  ssd_chunk_kernel<<<dim3(NCHUNK, NHEAD), 256, 0, stream>>>(
      Ch_bf, Bh_bf, Xh_bf, CumA, Ybuf, Sraw);
  scan_kernel<<<NHEAD, 256, 0, stream>>>(Sraw, Sused, CumA);
  yoff_kernel<<<dim3(NCHUNK, NHEAD), 256, 0, stream>>>(Ch_bf, Sused, CumA, Ybuf);

  z_kernel<<<(unsigned)(((size_t)NB * SEQ * 1024 + 255) / 256), 256, 0, stream>>>(
      Ybuf, NormB, Zbf);

  // out = Z @ W_out^T  (M=16384, N=1024, K=1024)
  gemm_bf16_wmma<<<dim3(NB * SEQ / 128, 1024 / 128), 256, 0, stream>>>(
      Zbf, WoutBf, out, NB * SEQ, 1024, 1024, 1024);
}
